// Pyramid_CrossattAlign_Atttrans_17755394801961
// MI455X (gfx1250) — matrix operand order repaired and checked by reading, not compiled, not run on hardware
//
#include <hip/hip_runtime.h>
#include <math.h>

typedef __bf16 bf16_t;
typedef __attribute__((ext_vector_type(16))) __bf16 v16bf;
typedef __attribute__((ext_vector_type(8)))  float  v8f;

#define LDS_LD 66      // padded row stride (elements) -> lane strides hit distinct LDS banks
#define CONV_LDP 102   // 10x10 halo patch + pad per channel

// ---------------------------------------------------------------------------
// WMMA helpers (CDNA5: D = A(16x32 bf16) x B(32x16 bf16) + C(16x16 f32))
// ---------------------------------------------------------------------------
__device__ __forceinline__ v8f wmma_bf16(v16bf a, v16bf b, v8f c) {
  return __builtin_amdgcn_wmma_f32_16x16x32_bf16(false, a, false, b, (short)0, c, false, false);
}

// A fragment from bf16 LDS, row-major [m][k]; zero-pads K >= kmax (for d=16 heads).
// Contiguous-k layout used here (B helpers match): lane m=lane&15, hi=lane>>4; VGPR r holds K pair hi*16+2*r.
__device__ __forceinline__ v16bf frag_a_lds(const bf16_t* p, int ld, int m0, int kbase,
                                            int kmax, int lane) {
  int m = m0 + (lane & 15);
  int hi = (lane >> 4) & 1;
  const bf16_t* row = p + m * ld + kbase;
  v16bf a;
#pragma unroll
  for (int r = 0; r < 8; ++r) {
    int kb = hi * 16 + 2 * r;
    a[2 * r]     = (kb < kmax)     ? row[kb]     : (bf16_t)0.0f;
    a[2 * r + 1] = (kb + 1 < kmax) ? row[kb + 1] : (bf16_t)0.0f;
  }
  return a;
}

// B fragment from bf16 LDS, row-major [k][n]. lanes 0-15: K=0..15, lanes 16-31: K=16..31.
__device__ __forceinline__ v16bf frag_b_lds(const bf16_t* p, int ld, int k0, int n0, int lane) {
  int n = n0 + (lane & 15);
  int hi = (lane >> 4) & 1;
  v16bf b;
#pragma unroll
  for (int r = 0; r < 8; ++r) {
    int k = k0 + hi * 16 + 2 * r;
    b[2 * r]     = p[k * ld + n];
    b[2 * r + 1] = p[(k + 1) * ld + n];
  }
  return b;
}

// B = K^T fragment: B[k][n] = p[n*ld + cbase + k] (token-major K matrix), zero past kmax.
__device__ __forceinline__ v16bf frag_bt_lds(const bf16_t* p, int ld, int cbase, int kmax,
                                             int n0, int lane) {
  int n = n0 + (lane & 15);
  int hi = (lane >> 4) & 1;
  const bf16_t* col = p + n * ld + cbase;
  v16bf b;
#pragma unroll
  for (int r = 0; r < 8; ++r) {
    int k = hi * 16 + 2 * r;
    b[2 * r]     = (k < kmax)     ? col[k]     : (bf16_t)0.0f;
    b[2 * r + 1] = (k + 1 < kmax) ? col[k + 1] : (bf16_t)0.0f;
  }
  return b;
}

// B fragment straight from global f32 weights, row-major [k][n] (small, L2-resident).
__device__ __forceinline__ v16bf frag_b_gf32(const float* p, int ld, int k0, int n0, int lane) {
  int n = n0 + (lane & 15);
  int hi = (lane >> 4) & 1;
  v16bf b;
#pragma unroll
  for (int r = 0; r < 8; ++r) {
    int k = k0 + hi * 16 + 2 * r;
    b[2 * r]     = (bf16_t)p[(size_t)k * ld + n];
    b[2 * r + 1] = (bf16_t)p[(size_t)(k + 1) * ld + n];
  }
  return b;
}

// Conv weight B fragment: Wc layout (O=64, I=192, 3, 3); B[k=cin][n=cout] at tap.
__device__ __forceinline__ v16bf frag_b_convw(const float* w, int k0, int n0, int tap, int lane) {
  int n = n0 + (lane & 15);
  int hi = (lane >> 4) & 1;
  const float* base = w + (size_t)n * (192 * 9) + tap;
  v16bf b;
#pragma unroll
  for (int r = 0; r < 8; ++r) {
    int k = k0 + hi * 16 + 2 * r;
    b[2 * r]     = (bf16_t)base[k * 9];
    b[2 * r + 1] = (bf16_t)base[(k + 1) * 9];
  }
  return b;
}

// Conv A fragment from halo patch: A[m=pixel][k=cin] shifted by tap (ky,kx).
__device__ __forceinline__ v16bf frag_a_conv(const bf16_t* sp, int m0, int k0,
                                             int ky, int kx, int lane) {
  int m = m0 + (lane & 15);
  int off = ((m >> 3) + ky) * 10 + (m & 7) + kx;
  int hi = (lane >> 4) & 1;
  v16bf a;
#pragma unroll
  for (int r = 0; r < 8; ++r) {
    int kb = k0 + hi * 16 + 2 * r;
    a[2 * r]     = sp[kb * CONV_LDP + off];
    a[2 * r + 1] = sp[(kb + 1) * CONV_LDP + off];
  }
  return a;
}

// C/D tile epilogue indices: n = n0+(lane&15), rows m0 + (lane>=16)*8 + r.

// ---------------------------------------------------------------------------
// Kernel 1: windowed cross-attention (one 8x8 window per 256-thread block)
// x = ref (masked -> Q), y = to (-> K,V). Outputs NCHW aligned + head-mean att.
// Dynamic LDS = 93,440 bytes.
// ---------------------------------------------------------------------------
#define ATTN_SMEM 93440

__global__ void __launch_bounds__(256) attn_win_kernel(
    const float* __restrict__ xg, const float* __restrict__ yg,
    const float* __restrict__ Wq, const float* __restrict__ bq,
    const float* __restrict__ Wkv, const float* __restrict__ bkv,
    const float* __restrict__ Wp, const float* __restrict__ bp,
    float* __restrict__ alignedOut, float* __restrict__ attOut, int S) {
  extern __shared__ char smem_raw[];
  float*  sS    = (float*)smem_raw;        // scores, one head at a time (f32)
  float*  sA    = sS + 64 * LDS_LD;        // head-mean attention accumulator
  float*  sMask = sA + 64 * LDS_LD;
  float*  sLog  = sMask + 64;
  bf16_t* sX = (bf16_t*)(sLog + 64);       // masked ref window
  bf16_t* sY = sX + 64 * LDS_LD;           // to window
  bf16_t* sQ = sY + 64 * LDS_LD;
  bf16_t* sK = sQ + 64 * LDS_LD;
  bf16_t* sV = sK + 64 * LDS_LD;
  bf16_t* sP = sV + 64 * LDS_LD;           // softmax probs (bf16)
  bf16_t* sO = sP + 64 * LDS_LD;           // attention output (pre-projection)

  const int tid = threadIdx.x, lane = tid & 31, wave = tid >> 5;
  const int nwx = S >> 3, winPerB = nwx * nwx;
  const int w = blockIdx.x, b = w / winPerB, wr = w - b * winPerB;
  const int wy = wr / nwx, wx = wr - (wr / nwx) * nwx;
  const size_t plane = (size_t)S * S;
  const size_t base = (size_t)b * 64 * plane + (size_t)(wy * 8) * S + (size_t)(wx * 8);

  // overexposure mask per token: mean_c(x > 0.95 ? 0 : 1)
  if (tid < 64) {
    const float* px = xg + base + (size_t)(tid >> 3) * S + (tid & 7);
    float cnt = 0.f;
#pragma unroll 4
    for (int c = 0; c < 64; ++c) cnt += (px[(size_t)c * plane] > 0.95f) ? 0.f : 1.f;
    float m = cnt * (1.0f / 64.0f);
    sMask[tid] = m;
    sLog[tid]  = __logf(m + 1e-6f);
  }
  __syncthreads();

  // stage windows (fp32 -> bf16 once)
  for (int i = tid; i < 64 * 64; i += 256) {
    int c = i >> 6, t = i & 63;
    size_t p = base + (size_t)c * plane + (size_t)(t >> 3) * S + (t & 7);
    sX[t * LDS_LD + c] = (bf16_t)(xg[p] * sMask[t]);
    sY[t * LDS_LD + c] = (bf16_t)yg[p];
  }
  __syncthreads();

  // Q = (sX @ Wq + bq) * d^-0.5     (M=64,N=64,K=64) -> 16 tiles over 8 waves
  for (int job = wave; job < 16; job += 8) {
    int m0 = (job >> 2) << 4, n0 = (job & 3) << 4;
    v8f acc = {0.f, 0.f, 0.f, 0.f, 0.f, 0.f, 0.f, 0.f};
    acc = wmma_bf16(frag_a_lds(sX, LDS_LD, m0, 0, 32, lane),  frag_b_gf32(Wq, 64, 0, n0, lane),  acc);
    acc = wmma_bf16(frag_a_lds(sX, LDS_LD, m0, 32, 32, lane), frag_b_gf32(Wq, 64, 32, n0, lane), acc);
    int n = n0 + (lane & 15);
    float bias = bq[n];
    int mb = m0 + ((lane >> 4) & 1) * 8;
#pragma unroll
    for (int r = 0; r < 8; ++r)
      sQ[(mb + r) * LDS_LD + n] = (bf16_t)((acc[r] + bias) * 0.25f);  // d=16 -> 1/sqrt(d)
  }
  // KV = sY @ Wkv + bkv   (N=128: cols 0..63 -> K, 64..127 -> V)
  for (int job = wave; job < 32; job += 8) {
    int m0 = (job >> 3) << 4, n0 = (job & 7) << 4;
    v8f acc = {0.f, 0.f, 0.f, 0.f, 0.f, 0.f, 0.f, 0.f};
    acc = wmma_bf16(frag_a_lds(sY, LDS_LD, m0, 0, 32, lane),  frag_b_gf32(Wkv, 128, 0, n0, lane),  acc);
    acc = wmma_bf16(frag_a_lds(sY, LDS_LD, m0, 32, 32, lane), frag_b_gf32(Wkv, 128, 32, n0, lane), acc);
    int n = n0 + (lane & 15);
    float bias = bkv[n];
    bf16_t* dst = (n < 64) ? sK : sV;
    int nn = n & 63;
    int mb = m0 + ((lane >> 4) & 1) * 8;
#pragma unroll
    for (int r = 0; r < 8; ++r)
      dst[(mb + r) * LDS_LD + nn] = (bf16_t)(acc[r] + bias);
  }
  for (int i = tid; i < 64 * LDS_LD; i += 256) sA[i] = 0.f;
  __syncthreads();

  for (int h = 0; h < 4; ++h) {
    const int cb = h << 4;
    // scores = Qh(64x16) @ Kh^T(16x64), K-dim zero-padded to 32 in fragments
    for (int job = wave; job < 16; job += 8) {
      int m0 = (job >> 2) << 4, n0 = (job & 3) << 4;
      v8f acc = {0.f, 0.f, 0.f, 0.f, 0.f, 0.f, 0.f, 0.f};
      acc = wmma_bf16(frag_a_lds(sQ, LDS_LD, m0, cb, 16, lane),
                      frag_bt_lds(sK, LDS_LD, cb, 16, n0, lane), acc);
      int n = n0 + (lane & 15);
      int mb = m0 + ((lane >> 4) & 1) * 8;
#pragma unroll
      for (int r = 0; r < 8; ++r) sS[(mb + r) * LDS_LD + n] = acc[r];
    }
    __syncthreads();
    // softmax over keys with log-mask bias; accumulate head-mean attention
    if (tid < 64) {
      float* row = sS + tid * LDS_LD;
      float mx = -1e30f;
#pragma unroll 4
      for (int j = 0; j < 64; ++j) { float v = row[j] + sLog[j]; row[j] = v; mx = fmaxf(mx, v); }
      float sum = 0.f;
#pragma unroll 4
      for (int j = 0; j < 64; ++j) { float e = __expf(row[j] - mx); row[j] = e; sum += e; }
      float inv = 1.0f / sum;
      bf16_t* prow = sP + tid * LDS_LD;
      float*  arow = sA + tid * LDS_LD;
#pragma unroll 4
      for (int j = 0; j < 64; ++j) {
        float pv = row[j] * inv;
        prow[j] = (bf16_t)pv;
        arow[j] += 0.25f * pv;
      }
    }
    __syncthreads();
    // out_h = P(64x64) @ V[:, cb:cb+16]
    for (int job = wave; job < 4; job += 8) {
      int m0 = job << 4;
      v8f acc = {0.f, 0.f, 0.f, 0.f, 0.f, 0.f, 0.f, 0.f};
      acc = wmma_bf16(frag_a_lds(sP, LDS_LD, m0, 0, 32, lane),  frag_b_lds(sV, LDS_LD, 0, cb, lane),  acc);
      acc = wmma_bf16(frag_a_lds(sP, LDS_LD, m0, 32, 32, lane), frag_b_lds(sV, LDS_LD, 32, cb, lane), acc);
      int n = cb + (lane & 15);
      int mb = m0 + ((lane >> 4) & 1) * 8;
#pragma unroll
      for (int r = 0; r < 8; ++r) sO[(mb + r) * LDS_LD + n] = (bf16_t)acc[r];
    }
    __syncthreads();
  }

  // projection + NCHW scatter: aligned = sO @ Wp + bp
  for (int job = wave; job < 16; job += 8) {
    int m0 = (job >> 2) << 4, n0 = (job & 3) << 4;
    v8f acc = {0.f, 0.f, 0.f, 0.f, 0.f, 0.f, 0.f, 0.f};
    acc = wmma_bf16(frag_a_lds(sO, LDS_LD, m0, 0, 32, lane),  frag_b_gf32(Wp, 64, 0, n0, lane),  acc);
    acc = wmma_bf16(frag_a_lds(sO, LDS_LD, m0, 32, 32, lane), frag_b_gf32(Wp, 64, 32, n0, lane), acc);
    int n = n0 + (lane & 15);
    float bias = bp[n];
    int mb = m0 + ((lane >> 4) & 1) * 8;
#pragma unroll
    for (int r = 0; r < 8; ++r) {
      int t = mb + r;
      alignedOut[base + (size_t)n * plane + (size_t)(t >> 3) * S + (t & 7)] = acc[r] + bias;
    }
  }
  if (attOut) {
    for (int i = tid; i < 64 * 64; i += 256)
      attOut[(size_t)w * 4096 + i] = sA[(i >> 6) * LDS_LD + (i & 63)];
  }
}

// ---------------------------------------------------------------------------
// Kernel 2: attention transfer: out = head_mean_att(64x64) @ rearranged(64x256)
// toa: NCHW at level i (wsz = 16, pr = 2). One 16x16 window per block.
// ---------------------------------------------------------------------------
__global__ void __launch_bounds__(256) atttrans_kernel(
    const float* __restrict__ toa, const float* __restrict__ att,
    float* __restrict__ outp, int S) {
  __shared__ bf16_t sAtt[64 * LDS_LD];
  __shared__ bf16_t sW[64 * 258];
  const int tid = threadIdx.x, lane = tid & 31, wave = tid >> 5;
  const int nwx = S >> 4, winPerB = nwx * nwx;
  const int w = blockIdx.x, b = w / winPerB, wr = w - b * winPerB;
  const int wy = wr / nwx, wx = wr % nwx;
  const size_t plane = (size_t)S * S;
  const size_t cbase = (size_t)b * 64 * plane;

  for (int i = tid; i < 4096; i += 256)
    sAtt[(i >> 6) * LDS_LD + (i & 63)] = (bf16_t)att[(size_t)w * 4096 + i];
  // w[n][j]: n=(y8,x8) coarse token, j=(py*2+px)*64+c fine offset+channel
  for (int i = tid; i < 64 * 256; i += 256) {
    int n = i >> 8, j = i & 255;
    int pp = j >> 6, c = j & 63, py = pp >> 1, px = pp & 1;
    int Y = wy * 16 + ((n >> 3) << 1) + py;
    int X = wx * 16 + ((n & 7) << 1) + px;
    sW[n * 258 + j] = (bf16_t)toa[cbase + (size_t)c * plane + (size_t)Y * S + X];
  }
  __syncthreads();

  for (int job = wave; job < 64; job += 8) {  // 4 m-tiles x 16 n-tiles
    int m0 = (job >> 4) << 4, n0 = (job & 15) << 4;
    v8f acc = {0.f, 0.f, 0.f, 0.f, 0.f, 0.f, 0.f, 0.f};
    acc = wmma_bf16(frag_a_lds(sAtt, LDS_LD, m0, 0, 32, lane),  frag_b_lds(sW, 258, 0, n0, lane),  acc);
    acc = wmma_bf16(frag_a_lds(sAtt, LDS_LD, m0, 32, 32, lane), frag_b_lds(sW, 258, 32, n0, lane), acc);
    int j = n0 + (lane & 15);
    int pp = j >> 6, c = j & 63, py = pp >> 1, px = pp & 1;
    int mb = m0 + ((lane >> 4) & 1) * 8;
#pragma unroll
    for (int r = 0; r < 8; ++r) {
      int t = mb + r;
      int Y = wy * 16 + ((t >> 3) << 1) + py;
      int X = wx * 16 + ((t & 7) << 1) + px;
      outp[cbase + (size_t)c * plane + (size_t)Y * S + X] = acc[r];
    }
  }
}

// ---------------------------------------------------------------------------
// Kernel 3: 3x3 SAME conv, 192 -> 64 ch, implicit GEMM. One 8x8 pixel tile
// per block; input = concat(inA, inB, inC) each 64ch NCHW. Optional lrelu.
// ---------------------------------------------------------------------------
__global__ void __launch_bounds__(256) conv3x3_kernel(
    const float* __restrict__ inA, const float* __restrict__ inB, const float* __restrict__ inC,
    const float* __restrict__ Wc, const float* __restrict__ bc,
    float* __restrict__ outp, int S, int doLrelu) {
  __shared__ bf16_t sPatch[192 * CONV_LDP];  // 10x10 halo per input channel
  const int tid = threadIdx.x, lane = tid & 31, wave = tid >> 5;
  const int tilesX = S >> 3, tilesPerB = tilesX * tilesX;
  const int blk = blockIdx.x, b = blk / tilesPerB, tr = blk - b * tilesPerB;
  const int by = (tr / tilesX) << 3, bx = (tr % tilesX) << 3;
  const size_t plane = (size_t)S * S;
  const size_t cbase = (size_t)b * 64 * plane;

  for (int i = tid; i < 192 * 100; i += 256) {
    int c = i / 100, pos = i - c * 100;
    int ly = pos / 10, lx = pos - ly * 10;
    int gy = by + ly - 1, gx = bx + lx - 1;
    float v = 0.f;
    if (gy >= 0 && gy < S && gx >= 0 && gx < S) {
      const float* src = (c < 64) ? inA : ((c < 128) ? inB : inC);
      v = src[cbase + (size_t)(c & 63) * plane + (size_t)gy * S + gx];
    }
    sPatch[c * CONV_LDP + ly * 10 + lx] = (bf16_t)v;
  }
  __syncthreads();

  for (int job = wave; job < 16; job += 8) {  // 4 pixel-tiles x 4 cout-tiles
    int m0 = (job >> 2) << 4, n0 = (job & 3) << 4;
    v8f acc = {0.f, 0.f, 0.f, 0.f, 0.f, 0.f, 0.f, 0.f};
    for (int tap = 0; tap < 9; ++tap) {
      int ky = tap / 3, kx = tap - ky * 3;
#pragma unroll
      for (int k0 = 0; k0 < 192; k0 += 32)
        acc = wmma_bf16(frag_a_conv(sPatch, m0, k0, ky, kx, lane),
                        frag_b_convw(Wc, k0, n0, tap, lane), acc);
    }
    int n = n0 + (lane & 15);
    float bias = bc[n];
    int mb = m0 + ((lane >> 4) & 1) * 8;
#pragma unroll
    for (int r = 0; r < 8; ++r) {
      float v = acc[r] + bias;
      if (doLrelu) v = (v > 0.f) ? v : 0.1f * v;
      int m = mb + r;
      outp[cbase + (size_t)n * plane + (size_t)(by + (m >> 3)) * S + (bx + (m & 7))] = v;
    }
  }
}

// ---------------------------------------------------------------------------
// Kernel 4: bilinear 2x upsample (half-pixel centers, edge clamp), optional
// lrelu applied to source samples (fuses reference's lrelu-then-resize).
// ---------------------------------------------------------------------------
__global__ void upsample2x_kernel(const float* __restrict__ inp, float* __restrict__ outp,
                                  int S, int lreluSrc) {
  const int So = S * 2;
  const size_t total = (size_t)8 * 64 * So * So;
  size_t idx = (size_t)blockIdx.x * blockDim.x + threadIdx.x;
  if (idx >= total) return;
  int x = (int)(idx % So);
  int y = (int)((idx / So) % So);
  size_t bcp = idx / ((size_t)So * So);
  float fy = (y + 0.5f) * 0.5f - 0.5f, fx = (x + 0.5f) * 0.5f - 0.5f;
  int y0 = (int)floorf(fy), x0 = (int)floorf(fx);
  float wy = fy - y0, wx = fx - x0;
  int y0c = y0 < 0 ? 0 : (y0 > S - 1 ? S - 1 : y0);
  int y1c = y0 + 1 < 0 ? 0 : (y0 + 1 > S - 1 ? S - 1 : y0 + 1);
  int x0c = x0 < 0 ? 0 : (x0 > S - 1 ? S - 1 : x0);
  int x1c = x0 + 1 < 0 ? 0 : (x0 + 1 > S - 1 ? S - 1 : x0 + 1);
  const float* pl = inp + bcp * (size_t)S * S;
  float v00 = pl[(size_t)y0c * S + x0c], v01 = pl[(size_t)y0c * S + x1c];
  float v10 = pl[(size_t)y1c * S + x0c], v11 = pl[(size_t)y1c * S + x1c];
  if (lreluSrc) {
    v00 = v00 > 0.f ? v00 : 0.1f * v00;
    v01 = v01 > 0.f ? v01 : 0.1f * v01;
    v10 = v10 > 0.f ? v10 : 0.1f * v10;
    v11 = v11 > 0.f ? v11 : 0.1f * v11;
  }
  outp[idx] = v00 * (1.f - wy) * (1.f - wx) + v01 * (1.f - wy) * wx +
              v10 * wy * (1.f - wx) + v11 * wy * wx;
}

// ---------------------------------------------------------------------------
// Host orchestration
// ---------------------------------------------------------------------------
extern "C" void kernel_launch(void* const* d_in, const int* in_sizes, int n_in,
                              void* d_out, int out_size, void* d_ws, size_t ws_size,
                              hipStream_t stream) {
  const float* ref1 = (const float*)d_in[0];
  const float* to1  = (const float*)d_in[1];
  const float* ref2 = (const float*)d_in[2];
  const float* to2  = (const float*)d_in[3];
  const float* ref3 = (const float*)d_in[4];
  const float* to3  = (const float*)d_in[5];
  const float *Wq1 = (const float*)d_in[6],  *bq1 = (const float*)d_in[7];
  const float *Wkv1 = (const float*)d_in[8], *bkv1 = (const float*)d_in[9];
  const float *Wp1 = (const float*)d_in[10], *bp1 = (const float*)d_in[11];
  const float *Wq2 = (const float*)d_in[12], *bq2 = (const float*)d_in[13];
  const float *Wkv2 = (const float*)d_in[14], *bkv2 = (const float*)d_in[15];
  const float *Wp2 = (const float*)d_in[16], *bp2 = (const float*)d_in[17];
  const float *Wq3 = (const float*)d_in[18], *bq3 = (const float*)d_in[19];
  const float *Wkv3 = (const float*)d_in[20], *bkv3 = (const float*)d_in[21];
  const float *Wp3 = (const float*)d_in[22], *bp3 = (const float*)d_in[23];
  const float *Wc1 = (const float*)d_in[24], *bc1 = (const float*)d_in[25];
  const float *Wc2 = (const float*)d_in[26], *bc2 = (const float*)d_in[27];

  const size_t L1 = (size_t)8 * 64 * 128 * 128;  // 8,388,608
  const size_t L2 = (size_t)8 * 64 * 64 * 64;    // 2,097,152
  const size_t L3 = (size_t)8 * 64 * 32 * 32;    //   524,288

  float* ws = (float*)d_ws;
  float* regA = ws;            // aligned2 / feat2 / aligned1
  float* regB = ws + L1;       // up3 / up2
  float* regC = ws + 2 * L1;   // feat3 / at2 / att3 / at1
  float* regD = ws + 3 * L1;   // att2

  float* feat3    = regC;
  float* at2      = regC + L3;
  float* att3     = regC + L3 + L2;
  float* at1      = regC;
  float* up3      = regB;
  float* up2      = regB;
  float* aligned2 = regA;
  float* feat2    = regA + L2;
  float* aligned1 = regA;
  float* att2     = regD;

  (void)hipFuncSetAttribute(reinterpret_cast<const void*>(attn_win_kernel),
                            hipFuncAttributeMaxDynamicSharedMemorySize, ATTN_SMEM);

  // ---- level 3 (coarsest, S=32) ----
  attn_win_kernel<<<128, 256, ATTN_SMEM, stream>>>(
      ref3, to3, Wq3, bq3, Wkv3, bkv3, Wp3, bp3, feat3, att3, 32);
  upsample2x_kernel<<<(int)(((size_t)8 * 64 * 64 * 64 + 255) / 256), 256, 0, stream>>>(
      feat3, up3, 32, /*lreluSrc=*/1);

  // ---- level 2 (S=64) ----
  attn_win_kernel<<<512, 256, ATTN_SMEM, stream>>>(
      ref2, to2, Wq2, bq2, Wkv2, bkv2, Wp2, bp2, aligned2, att2, 64);
  atttrans_kernel<<<128, 256, 0, stream>>>(to2, att3, at2, 64);
  conv3x3_kernel<<<512, 256, 0, stream>>>(aligned2, up3, at2, Wc2, bc2, feat2, 64,
                                          /*doLrelu=*/1);
  upsample2x_kernel<<<(int)((L1 + 255) / 256), 256, 0, stream>>>(feat2, up2, 64,
                                                                 /*lreluSrc=*/0);

  // ---- level 1 (finest, S=128) ----
  attn_win_kernel<<<2048, 256, ATTN_SMEM, stream>>>(
      ref1, to1, Wq1, bq1, Wkv1, bkv1, Wp1, bp1, aligned1, nullptr, 128);
  atttrans_kernel<<<512, 256, 0, stream>>>(to1, att2, at1, 128);
  conv3x3_kernel<<<2048, 256, 0, stream>>>(aligned1, up2, at1, Wc1, bc1,
                                           (float*)d_out, 128, /*doLrelu=*/1);
}
